// _BaseSpiderConv_81415400063748
// MI455X (gfx1250) — compile-verified, matrix operand order checked by reading, not run on hardware
//
#include <hip/hip_runtime.h>

// CDNA5 / gfx1250 wave32 WMMA implementation of BaseSpiderConv.
// The (c,t,k) contraction is a 32768 x 3840 x 64 GEMM in f16 with f32
// accumulation (v_wmma_f32_16x16x32_f16). A-matrix (gathered feat * taylor)
// is staged in LDS as packed f16; conv_w is pre-swizzled into WMMA B-fragment
// order and DMA'd into LDS per k-step with the Tensor Data Mover
// (tensor_load_to_lds), shared by all 8 waves of the block.

typedef __attribute__((ext_vector_type(16))) _Float16 v16h;
typedef __attribute__((ext_vector_type(2)))  _Float16 h2;
typedef __attribute__((ext_vector_type(8)))  float    v8f;
typedef __attribute__((ext_vector_type(4)))  unsigned su4;
typedef __attribute__((ext_vector_type(8)))  unsigned su8;
typedef __attribute__((ext_vector_type(4)))  unsigned u4v;

#define B_  8
#define C_  64
#define N_  4096
#define K_  20
#define T_  3
#define O_  64
#define P_  20
#define CTK_ 3840      /* C*T*K */
#define NCHUNK_ 120    /* CTK/32 */
#define MTOT_ 32768    /* B*N */
#define BSLAB_ 6144    /* dwords of B-fragments per k step (6 chunks * 1024) */

// ---------------------------------------------------------------- k_zero ----
__global__ void k_zero(float* __restrict__ p, int n) {
    int i = blockIdx.x * blockDim.x + threadIdx.x;
    if (i < n) p[i] = 0.0f;
}

// ----------------------------------------------------------- k_transpose ----
// feat (B,C,N) -> featT (B,N,C) so gathered rows are contiguous 64 floats.
__global__ void k_transpose(const float* __restrict__ feat,
                            float* __restrict__ featT) {
    int i = blockIdx.x * 256 + threadIdx.x;   // B*C*N = 2,097,152
    int b = i >> 18;                          // C*N = 2^18
    int r = i & 262143;
    int c = r >> 12;                          // N = 2^12
    int n = r & 4095;
    featT[(((b << 12) | n) << 6) + c] = feat[i];
}

// -------------------------------------------------------------- k_taylor ----
// taylor[b,t,n,k] = sum_p weights[b,p,t,n,k] * poly_terms(group_pc)[p]
__global__ void k_taylor(const float* __restrict__ gpc,
                         const float* __restrict__ w,
                         float* __restrict__ ty) {
    int i = blockIdx.x * 256 + threadIdx.x;   // B*N*K = 655,360
    int k = i % K_;
    int n = (i / K_) % N_;
    int b = i / (N_ * K_);

    float X = gpc[((b * 3 + 0) * N_ + n) * K_ + k];
    float Y = gpc[((b * 3 + 1) * N_ + n) * K_ + k];
    float Z = gpc[((b * 3 + 2) * N_ + n) * K_ + k];
    float XX = X * X, YY = Y * Y, ZZ = Z * Z;
    float XY = X * Y, XZ = X * Z, YZ = Y * Z;
    float term[P_] = {
        1.0f, X, Y, Z, XX, YY, ZZ,
        XX * X, YY * Y, ZZ * Z,
        XY, XZ, YZ,
        X * XY, X * XZ, Y * YZ, Y * XY, Z * XZ, Z * YZ, XY * Z
    };
    #pragma unroll
    for (int t = 0; t < T_; ++t) {
        float acc = 0.0f;
        #pragma unroll
        for (int p = 0; p < P_; ++p) {
            acc += w[(((b * 20 + p) * 3 + t) * N_ + n) * K_ + k] * term[p];
        }
        ty[((b * 3 + t) * N_ + n) * K_ + k] = acc;
    }
}

// -------------------------------------------------------------- k_repack ----
// conv_w (O, C*T, K) -> f16 B-fragments in WMMA operand order, stored so each
// lane's 8 dwords are contiguous:  bfrag[((chunk*4 + ot)*32 + lane)*8 + vgpr]
// packs two f16 (even K index in [15:0]). chunk = (k*T + t)*2 + chalf, with c
// innermost in the contraction so one taylor scalar covers a 32-wide chunk.
__global__ void k_repack(const float* __restrict__ cw,
                         unsigned* __restrict__ bfrag) {
    int tid   = blockIdx.x * 256 + threadIdx.x;  // 120*4*32*8 = 122,880
    int v     = tid & 7;
    int lane  = (tid >> 3) & 31;
    int ot    = (tid >> 8) & 3;
    int chunk = tid >> 10;                       // 0..119
    int ch    = chunk & 1;
    int t     = (chunk >> 1) % 3;
    int k     = chunk / 6;
    int half  = lane >> 4;
    int nn    = lane & 15;
    int o     = ot * 16 + nn;
    // ISA 16-bit operand layout: lane-half h holds K = {8h+0..7, 16+8h+0..7}
    int kk    = (v < 4) ? (8 * half + 2 * v) : (16 + 8 * half + 2 * (v - 4));
    int c     = ch * 32 + kk;
    float f0 = cw[o * CTK_ + (c * T_ + t) * K_ + k];
    float f1 = cw[o * CTK_ + ((c + 1) * T_ + t) * K_ + k];
    bfrag[tid] = __builtin_bit_cast(unsigned,
                     __builtin_amdgcn_cvt_pkrtz(f0, f1));
}

// ---------------------------------------------------------------- k_gemm ----
// 256 blocks x 256 threads (8 waves). Each wave owns a 16-row M tile and all
// 64 output channels (4 v8f accumulators) -> 480 WMMAs per wave. Per k step,
// wave 0 DMAs the 24KB B-fragment slab into LDS with the TDM while all waves
// stage their gathered feat rows (converted to packed f16); then 24 WMMAs run
// entirely out of LDS with only v_pk_mul_f16 on the A path.
__global__ void __launch_bounds__(256) k_gemm(
    const float* __restrict__ featT, const int* __restrict__ idx,
    const float* __restrict__ taylor, const unsigned* __restrict__ bfrag,
    float* __restrict__ out, float* __restrict__ sums,
    float* __restrict__ sumsq) {

    // packed-f16 gathered feat: row stride 36 dwords (144B: 16B aligned,
    // m*36 mod 64 -> 16 distinct banks).  18.4 KB
    __shared__ unsigned gfh[8][16][36];
    __shared__ unsigned lds_b[BSLAB_];    // B-fragment slab (24.0 KB)

    const int lane = threadIdx.x & 31;
    const int wave = threadIdx.x >> 5;
    const int half = lane >> 4;
    const int mrow = lane & 15;

    const int row0  = blockIdx.x * 128 + wave * 16;  // block stays in one b
    const int b     = row0 >> 12;
    const int nbase = row0 & 4095;
    const int n_m   = nbase + mrow;

    const float* fb = featT + ((long)b << 12) * 64;
    const unsigned ldsb_addr = (unsigned)(size_t)(&lds_b[0]); // LDS byte addr

    v8f acc[4];
    #pragma unroll
    for (int ot = 0; ot < 4; ++ot) acc[ot] = (v8f){};

    for (int k = 0; k < K_; ++k) {
        __syncthreads();   // everyone done reading previous B slab

        if (wave == 0) {
            // ---- Tensor Data Mover: 24KB contiguous slab -> LDS ----------
            unsigned long long ga =
                (unsigned long long)(size_t)(bfrag + k * BSLAB_);
            su4 g0;
            g0[0] = 1u;                                  // count=1, user mode
            g0[1] = ldsb_addr;                           // lds_addr
            g0[2] = (unsigned)ga;                        // global_addr[31:0]
            g0[3] = ((unsigned)(ga >> 32) & 0x1FFFFFFu)  // global_addr[56:32]
                    | 0x80000000u;                       // type=2 (image)
            su8 g1;
            g1[0] = 0x20000u;      // data_size=4B, no multicast/pad/iterate
            g1[1] = (unsigned)(BSLAB_ & 0xFFFF) << 16;   // tensor_dim0 lo16
            g1[2] = 0x10000u;                            // tensor_dim1 = 1
            g1[3] = (unsigned)(BSLAB_ & 0xFFFF) << 16;   // tile_dim0 = 6144
            g1[4] = 1u;                                  // tile_dim1 = 1
            g1[5] = (unsigned)BSLAB_;                    // dim0 stride
            g1[6] = 0u;
            g1[7] = 0u;
            asm volatile("tensor_load_to_lds %0, %1"
                         :: "s"(g0), "s"(g1) : "memory");
        }

        // ---- gather-stage feat row as packed f16 (overlaps the TDM) ------
        const int j = idx[(b * N_ + n_m) * K_ + k];
        const float4* src = (const float4*)(fb + j * 64 + half * 32);
        unsigned* dstrow = &gfh[wave][mrow][half * 16];
        #pragma unroll
        for (int q = 0; q < 4; ++q) {
            float4 f0 = src[2 * q];
            float4 f1 = src[2 * q + 1];
            u4v pk;
            pk[0] = __builtin_bit_cast(unsigned,
                        __builtin_amdgcn_cvt_pkrtz(f0.x, f0.y));
            pk[1] = __builtin_bit_cast(unsigned,
                        __builtin_amdgcn_cvt_pkrtz(f0.z, f0.w));
            pk[2] = __builtin_bit_cast(unsigned,
                        __builtin_amdgcn_cvt_pkrtz(f1.x, f1.y));
            pk[3] = __builtin_bit_cast(unsigned,
                        __builtin_amdgcn_cvt_pkrtz(f1.z, f1.w));
            *(u4v*)(dstrow + 4 * q) = pk;
        }

        const float ty0 = taylor[((b * 3 + 0) * N_ + n_m) * K_ + k];
        const float ty1 = taylor[((b * 3 + 1) * N_ + n_m) * K_ + k];
        const float ty2 = taylor[((b * 3 + 2) * N_ + n_m) * K_ + k];
        const float tys[3] = {ty0, ty1, ty2};

        if (wave == 0) __builtin_amdgcn_s_wait_tensorcnt(0);
        __syncthreads();   // B slab visible to all waves

        const unsigned* rowp = &gfh[wave][mrow][0];

        #pragma unroll
        for (int t = 0; t < 3; ++t) {
            const _Float16 tvh = (_Float16)tys[t];
            h2 tv2; tv2[0] = tvh; tv2[1] = tvh;
            #pragma unroll
            for (int ch = 0; ch < 2; ++ch) {
                // f16 base index = ch*32 + half*8  ->  dword idx /2
                const int dbase = ch * 16 + half * 4;
                union { u4v q[2]; h2 p[8]; v16h v; } au;
                au.q[0] = *(const u4v*)(rowp + dbase);
                au.q[1] = *(const u4v*)(rowp + dbase + 8);
                #pragma unroll
                for (int i = 0; i < 8; ++i) au.p[i] = au.p[i] * tv2;

                const int lc = t * 2 + ch;   // local chunk 0..5
                #pragma unroll
                for (int ot = 0; ot < 4; ++ot) {
                    const u4v* bp =
                        (const u4v*)&lds_b[(((lc * 4 + ot) * 32) + lane) * 8];
                    union { u4v q[2]; v16h v; } bu;
                    bu.q[0] = bp[0];
                    bu.q[1] = bp[1];
                    acc[ot] = __builtin_amdgcn_wmma_f32_16x16x32_f16(
                        false, au.v, false, bu.v, (short)0, acc[ot],
                        false, false);
                }
            }
        }
    }

    // ---- epilogue: C/D layout lane = o column, vgpr+8*half = m row --------
    const int ocol = lane & 15;
    #pragma unroll
    for (int ot = 0; ot < 4; ++ot) {
        const int o = ot * 16 + ocol;
        float s = 0.0f, ss = 0.0f;
        #pragma unroll
        for (int v = 0; v < 8; ++v) {
            float x = acc[ot][v];
            s  += x;
            ss += x * x;
            int n = nbase + half * 8 + v;
            out[((b * 64 + o) << 12) + n] = x;
        }
        atomicAdd(&sums[o], s);
        atomicAdd(&sumsq[o], ss);
    }
}

// ------------------------------------------------------------------ k_bn ----
__global__ void k_bn(float* __restrict__ out,
                     const float* __restrict__ sums,
                     const float* __restrict__ sumsq,
                     const float* __restrict__ gamma,
                     const float* __restrict__ beta) {
    int i = blockIdx.x * 256 + threadIdx.x;   // B*O*N = 2,097,152
    int o = (i >> 12) & 63;
    const float cinv = 1.0f / (float)MTOT_;
    float mean = sums[o] * cinv;
    float var  = sumsq[o] * cinv - mean * mean;
    float sc   = gamma[o] * rsqrtf(var + 1e-5f);
    float y    = (out[i] - mean) * sc + beta[o];
    out[i] = fmaxf(y, 0.0f);
}

// -------------------------------------------------------------- launcher ----
extern "C" void kernel_launch(void* const* d_in, const int* in_sizes, int n_in,
                              void* d_out, int out_size, void* d_ws,
                              size_t ws_size, hipStream_t stream) {
    const float* feat    = (const float*)d_in[0];
    const int*   idx     = (const int*)  d_in[1];
    const float* gpc     = (const float*)d_in[2];
    const float* weights = (const float*)d_in[3];
    const float* conv_w  = (const float*)d_in[4];
    const float* gamma   = (const float*)d_in[5];
    const float* beta    = (const float*)d_in[6];
    float* out = (float*)d_out;

    // workspace layout (floats)
    float*    featT  = (float*)d_ws;                        // 2,097,152
    float*    taylor = featT + 2097152;                     // 1,966,080
    unsigned* bfrag  = (unsigned*)(taylor + 1966080);       //   122,880
    float*    sums   = (float*)(bfrag + 122880);            //        64
    float*    sumsq  = sums + 64;                           //        64

    k_zero     <<<1, 128, 0, stream>>>(sums, 128);
    k_transpose<<<8192, 256, 0, stream>>>(feat, featT);
    k_taylor   <<<2560, 256, 0, stream>>>(gpc, weights, taylor);
    k_repack   <<<480, 256, 0, stream>>>(conv_w, bfrag);
    k_gemm     <<<256, 256, 0, stream>>>(featT, idx, taylor, bfrag,
                                         out, sums, sumsq);
    k_bn       <<<8192, 256, 0, stream>>>(out, sums, sumsq, gamma, beta);
}
